// SelfAttention_42949672960506
// MI455X (gfx1250) — matrix operand order repaired and checked by reading, not compiled, not run on hardware
//
#include <hip/hip_runtime.h>

typedef __attribute__((ext_vector_type(16))) _Float16 v16h;
typedef __attribute__((ext_vector_type(8)))  _Float16 v8h;
typedef __attribute__((ext_vector_type(8)))  float    v8f;

#define B_    4
#define S_    2048
#define C_    1024
#define H_    16
#define CH_   64
#define QKVN  3072
#define ROWS_ (B_ * S_)   // 8192

// ---------------------------------------------------------------------------
// WMMA helpers (gfx1250: D = A(16x32 f16) x B(32x16 f16) + C(16x16 f32))
// ---------------------------------------------------------------------------
__device__ __forceinline__ v8f wmma16(v16h a, v16h b, v8f c) {
  return __builtin_amdgcn_wmma_f32_16x16x32_f16(false, a, false, b, (short)0, c,
                                                false, false);
}

// A-fragment: 16x32 tile, row-major with leading dim `ld` (halves).
// ISA layout: lanes 0-15 row M=lane, K = {0..7, 16..23}; lanes 16-31 same row,
// K = {8..15, 24..31}.  => two 16B loads at K-offsets ko and ko+16.
__device__ __forceinline__ v16h load_a16(const _Float16* base, int ld) {
  const int lane = threadIdx.x & 31;
  const int r    = lane & 15;
  const int ko   = (lane >> 4) << 3;                  // 0 or 8
  const _Float16* p = base + (size_t)r * ld + ko;
  v8h lo = *(const v8h*)(p);
  v8h hi = *(const v8h*)(p + 16);
  return __builtin_shufflevector(lo, hi, 0,1,2,3,4,5,6,7,8,9,10,11,12,13,14,15);
}

// B-fragment from Bt stored N x K row-major (column n of B == row n of Bt).
// ISA layout: lanes 0-15 col N=lane, K = {0..7, 16..23}; lanes 16-31 same col,
// K = {8..15, 24..31}.  => two 16B loads at K-offsets ko and ko+16.
__device__ __forceinline__ v16h load_b16(const _Float16* base, int ld) {
  const int lane = threadIdx.x & 31;
  const int c    = lane & 15;
  const int ko   = (lane >> 4) << 3;                  // 0 or 8
  const _Float16* p = base + (size_t)c * ld + ko;
  v8h lo = *(const v8h*)(p);
  v8h hi = *(const v8h*)(p + 16);
  return __builtin_shufflevector(lo, hi, 0,1,2,3,4,5,6,7,8,9,10,11,12,13,14,15);
}

__device__ __forceinline__ v8f v8f_zero() {
  v8f r;
#pragma unroll
  for (int i = 0; i < 8; ++i) r[i] = 0.f;
  return r;
}

// Reductions across a 16-lane half (a full row of a 16x16 C tile).
__device__ __forceinline__ float rowmax16(float v) {
  v = fmaxf(v, __shfl_xor(v, 1, 32));
  v = fmaxf(v, __shfl_xor(v, 2, 32));
  v = fmaxf(v, __shfl_xor(v, 4, 32));
  v = fmaxf(v, __shfl_xor(v, 8, 32));
  return v;
}
__device__ __forceinline__ float rowsum16(float v) {
  v += __shfl_xor(v, 1, 32);
  v += __shfl_xor(v, 2, 32);
  v += __shfl_xor(v, 4, 32);
  v += __shfl_xor(v, 8, 32);
  return v;
}

// ---------------------------------------------------------------------------
// Kernel 1: f32 -> f16 transposed weight conversion (wT[n][k] = w[k][n])
// ---------------------------------------------------------------------------
__global__ void __launch_bounds__(256)
convert_w_kernel(const float* __restrict__ qkv_w, const float* __restrict__ out_w,
                 _Float16* __restrict__ wqkvT, _Float16* __restrict__ woutT) {
  const int idx = blockIdx.x * 256 + threadIdx.x;        // exactly 4M threads
  if (idx < QKVN * C_) {
    const int n = idx >> 10;          // 0..3071
    const int k = idx & 1023;
    wqkvT[idx] = (_Float16)qkv_w[(size_t)k * QKVN + n];
  } else {
    const int j = idx - QKVN * C_;
    const int n = j >> 10;
    const int k = j & 1023;
    woutT[j] = (_Float16)out_w[(size_t)k * C_ + n];
  }
}

// ---------------------------------------------------------------------------
// Kernel 2: LayerNorm (one row of 1024 per 256-thread block) -> f16 h
// ---------------------------------------------------------------------------
__global__ void __launch_bounds__(256)
ln_kernel(const float* __restrict__ x, const float* __restrict__ lnw,
          const float* __restrict__ lnb, _Float16* __restrict__ h16) {
  const int row = blockIdx.x;
  const int tid = threadIdx.x;
  const float* xr = x + (size_t)row * C_;
  float v[4];
  float s = 0.f, ss = 0.f;
#pragma unroll
  for (int j = 0; j < 4; ++j) {
    v[j] = xr[tid + j * 256];
    s  += v[j];
    ss += v[j] * v[j];
  }
#pragma unroll
  for (int m = 16; m >= 1; m >>= 1) {
    s  += __shfl_xor(s,  m, 32);
    ss += __shfl_xor(ss, m, 32);
  }
  __shared__ float red[2][8];
  if ((tid & 31) == 0) { red[0][tid >> 5] = s; red[1][tid >> 5] = ss; }
  __syncthreads();
  s = 0.f; ss = 0.f;
#pragma unroll
  for (int w = 0; w < 8; ++w) { s += red[0][w]; ss += red[1][w]; }
  const float mu  = s * (1.f / C_);
  const float var = ss * (1.f / C_) - mu * mu;
  const float inv = rsqrtf(var + 1e-5f);
#pragma unroll
  for (int j = 0; j < 4; ++j) {
    const int c = tid + j * 256;
    h16[(size_t)row * C_ + c] = (_Float16)(((v[j] - mu) * inv) * lnw[c] + lnb[c]);
  }
}

// ---------------------------------------------------------------------------
// Kernel 3: QKV GEMM  (8192 x 3072, K=1024), 64x64 tile per wave.
// Epilogue scatters to q[B,H,S,64], k[B,H,S,64], vT[B,H,64,S] in f16.
// ---------------------------------------------------------------------------
__global__ void __launch_bounds__(256)
qkv_gemm_kernel(const _Float16* __restrict__ h16, const _Float16* __restrict__ wT,
                _Float16* __restrict__ q16, _Float16* __restrict__ k16,
                _Float16* __restrict__ vT16) {
  const int wave = threadIdx.x >> 5;
  const int lane = threadIdx.x & 31;
  const int tile = blockIdx.x * 8 + wave;     // 6144 tiles total
  const int mT = tile & 127;                  // 128 M-tiles of 64
  const int nT = tile >> 7;                   // 48  N-tiles of 64
  const int m0 = mT * 64, n0 = nT * 64;

  v8f acc[4][4];
#pragma unroll
  for (int i = 0; i < 4; ++i)
#pragma unroll
    for (int j = 0; j < 4; ++j) acc[i][j] = v8f_zero();

#pragma unroll 1
  for (int kt = 0; kt < C_ / 32; ++kt) {
    v16h a[4], b[4];
#pragma unroll
    for (int mt = 0; mt < 4; ++mt)
      a[mt] = load_a16(h16 + (size_t)(m0 + mt * 16) * C_ + kt * 32, C_);
#pragma unroll
    for (int nt = 0; nt < 4; ++nt)
      b[nt] = load_b16(wT + (size_t)(n0 + nt * 16) * C_ + kt * 32, C_);
#pragma unroll
    for (int mt = 0; mt < 4; ++mt)
#pragma unroll
      for (int nt = 0; nt < 4; ++nt)
        acc[mt][nt] = wmma16(a[mt], b[nt], acc[mt][nt]);
  }

  const int hi8 = (lane >> 4) << 3;
  const int col = lane & 15;
#pragma unroll
  for (int mt = 0; mt < 4; ++mt)
#pragma unroll
    for (int nt = 0; nt < 4; ++nt)
#pragma unroll
      for (int i = 0; i < 8; ++i) {
        const int m = m0 + mt * 16 + hi8 + i;
        const int n = n0 + nt * 16 + col;
        const int bb = m >> 11;               // batch
        const int s  = m & 2047;              // seq
        const int head = n / 192;
        const int w    = n - head * 192;
        const _Float16 val = (_Float16)acc[mt][nt][i];
        const size_t bhs = ((size_t)(bb * H_ + head) * S_ + s);
        if (w < 64)        q16[bhs * 64 + w] = val;
        else if (w < 128)  k16[bhs * 64 + (w - 64)] = val;
        else vT16[((size_t)(bb * H_ + head) * 64 + (w - 128)) * S_ + s] = val;
      }
}

// ---------------------------------------------------------------------------
// Kernel 4: flash attention.  One wave = 32 query rows of one (b,h).
// 64 key blocks of 32; online softmax; P tile bounced through LDS to convert
// C-layout -> A-layout for the PV WMMA.
// ---------------------------------------------------------------------------
#define PSTR 40   // LDS row stride in halves (80B: 16B-aligned, staggered banks)

__global__ void __launch_bounds__(256)
attn_kernel(const _Float16* __restrict__ q16, const _Float16* __restrict__ k16,
            const _Float16* __restrict__ vT16, _Float16* __restrict__ o16) {
  __shared__ _Float16 plds[8][32 * PSTR];
  const int wave = threadIdx.x >> 5;
  const int lane = threadIdx.x & 31;
  const int task = blockIdx.x * 8 + wave;     // 4096 tasks
  const int qb = task & 63;                   // 64 q-blocks of 32 rows
  const int bh = task >> 6;                   // 0..63
  const _Float16* Q  = q16  + ((size_t)bh * S_ + qb * 32) * CH_;
  const _Float16* K  = k16  + (size_t)bh * S_ * CH_;
  const _Float16* Vt = vT16 + (size_t)bh * CH_ * S_;

  v16h qf[2][2];
#pragma unroll
  for (int mt = 0; mt < 2; ++mt)
#pragma unroll
    for (int ks = 0; ks < 2; ++ks)
      qf[mt][ks] = load_a16(Q + (size_t)(mt * 16) * CH_ + ks * 32, CH_);

  v8f oacc[2][4];
#pragma unroll
  for (int mt = 0; mt < 2; ++mt)
#pragma unroll
    for (int nt = 0; nt < 4; ++nt) oacc[mt][nt] = v8f_zero();
  float mr[2][8], lr[2][8];
#pragma unroll
  for (int mt = 0; mt < 2; ++mt)
#pragma unroll
    for (int i = 0; i < 8; ++i) { mr[mt][i] = -1e30f; lr[mt][i] = 0.f; }

  const float scale = 0.03125f;               // 1/sqrt(1024)
  const int hi8 = (lane >> 4) << 3;
  const int col = lane & 15;

#pragma unroll 1
  for (int kb = 0; kb < S_ / 32; ++kb) {
    // ---- energy = Q Kt for this 32-key block --------------------------------
    v16h kf[2][2];
#pragma unroll
    for (int nt = 0; nt < 2; ++nt)
#pragma unroll
      for (int ks = 0; ks < 2; ++ks)
        kf[nt][ks] = load_b16(K + (size_t)(kb * 32 + nt * 16) * CH_ + ks * 32, CH_);
    v8f e[2][2];
#pragma unroll
    for (int mt = 0; mt < 2; ++mt)
#pragma unroll
      for (int nt = 0; nt < 2; ++nt) {
        e[mt][nt] = v8f_zero();
#pragma unroll
        for (int ks = 0; ks < 2; ++ks)
          e[mt][nt] = wmma16(qf[mt][ks], kf[nt][ks], e[mt][nt]);
      }

    // ---- online softmax over the 32 new columns -----------------------------
#pragma unroll
    for (int mt = 0; mt < 2; ++mt) {
      float alpha[8];
#pragma unroll
      for (int i = 0; i < 8; ++i) {
        const float e0 = e[mt][0][i] * scale;
        const float e1 = e[mt][1][i] * scale;
        const float mx   = rowmax16(fmaxf(e0, e1));
        const float mnew = fmaxf(mr[mt][i], mx);
        alpha[i] = __expf(mr[mt][i] - mnew);
        mr[mt][i] = mnew;
        const float p0 = __expf(e0 - mnew);
        const float p1 = __expf(e1 - mnew);
        e[mt][0][i] = p0;
        e[mt][1][i] = p1;
        lr[mt][i] = lr[mt][i] * alpha[i] + rowsum16(p0 + p1);
      }
#pragma unroll
      for (int nt = 0; nt < 4; ++nt)
#pragma unroll
        for (int i = 0; i < 8; ++i) oacc[mt][nt][i] *= alpha[i];
      // store P tile (C layout) to LDS as f16, row-major 32 x PSTR
      _Float16* pl = &plds[wave][0];
#pragma unroll
      for (int i = 0; i < 8; ++i) {
        pl[(mt * 16 + hi8 + i) * PSTR + col]      = (_Float16)e[mt][0][i];
        pl[(mt * 16 + hi8 + i) * PSTR + 16 + col] = (_Float16)e[mt][1][i];
      }
    }
    asm volatile("s_wait_dscnt 0" ::: "memory");

    // ---- O += P x V ---------------------------------------------------------
    v16h pf[2];
#pragma unroll
    for (int mt = 0; mt < 2; ++mt)
      pf[mt] = load_a16(&plds[wave][mt * 16 * PSTR], PSTR);
    v16h vf[4];
#pragma unroll
    for (int nt = 0; nt < 4; ++nt)
      vf[nt] = load_b16(Vt + (size_t)(nt * 16) * S_ + kb * 32, S_);
#pragma unroll
    for (int mt = 0; mt < 2; ++mt)
#pragma unroll
      for (int nt = 0; nt < 4; ++nt)
        oacc[mt][nt] = wmma16(pf[mt], vf[nt], oacc[mt][nt]);
  }

  // ---- normalize and store o as f16 [B*S, 1024] -----------------------------
  const int b = bh >> 4, h = bh & 15;
#pragma unroll
  for (int mt = 0; mt < 2; ++mt)
#pragma unroll
    for (int nt = 0; nt < 4; ++nt)
#pragma unroll
      for (int i = 0; i < 8; ++i) {
        const int row = qb * 32 + mt * 16 + hi8 + i;
        const float val = oacc[mt][nt][i] / lr[mt][i];
        o16[((size_t)(b * S_ + row)) * C_ + h * 64 + nt * 16 + col] = (_Float16)val;
      }
}

// ---------------------------------------------------------------------------
// Kernel 5: out projection (8192 x 1024, K=1024) + bias + residual, f32 out.
// ---------------------------------------------------------------------------
__global__ void __launch_bounds__(256)
out_gemm_kernel(const _Float16* __restrict__ o16, const _Float16* __restrict__ wT,
                const float* __restrict__ out_b, const float* __restrict__ x,
                float* __restrict__ out) {
  const int wave = threadIdx.x >> 5;
  const int lane = threadIdx.x & 31;
  const int tile = blockIdx.x * 8 + wave;     // 2048 tiles
  const int mT = tile & 127;
  const int nT = tile >> 7;                   // 16 N-tiles of 64
  const int m0 = mT * 64, n0 = nT * 64;

  v8f acc[4][4];
#pragma unroll
  for (int i = 0; i < 4; ++i)
#pragma unroll
    for (int j = 0; j < 4; ++j) acc[i][j] = v8f_zero();

#pragma unroll 1
  for (int kt = 0; kt < C_ / 32; ++kt) {
    v16h a[4], b[4];
#pragma unroll
    for (int mt = 0; mt < 4; ++mt)
      a[mt] = load_a16(o16 + (size_t)(m0 + mt * 16) * C_ + kt * 32, C_);
#pragma unroll
    for (int nt = 0; nt < 4; ++nt)
      b[nt] = load_b16(wT + (size_t)(n0 + nt * 16) * C_ + kt * 32, C_);
#pragma unroll
    for (int mt = 0; mt < 4; ++mt)
#pragma unroll
      for (int nt = 0; nt < 4; ++nt)
        acc[mt][nt] = wmma16(a[mt], b[nt], acc[mt][nt]);
  }

  const int hi8 = (lane >> 4) << 3;
  const int col = lane & 15;
#pragma unroll
  for (int mt = 0; mt < 4; ++mt)
#pragma unroll
    for (int nt = 0; nt < 4; ++nt)
#pragma unroll
      for (int i = 0; i < 8; ++i) {
        const int m = m0 + mt * 16 + hi8 + i;
        const int n = n0 + nt * 16 + col;
        out[(size_t)m * C_ + n] = acc[mt][nt][i] + out_b[n] + x[(size_t)m * C_ + n];
      }
}

// ---------------------------------------------------------------------------
extern "C" void kernel_launch(void* const* d_in, const int* in_sizes, int n_in,
                              void* d_out, int out_size, void* d_ws, size_t ws_size,
                              hipStream_t stream) {
  const float* x     = (const float*)d_in[0];
  const float* ln_w  = (const float*)d_in[1];
  const float* ln_b  = (const float*)d_in[2];
  const float* qkv_w = (const float*)d_in[3];
  const float* out_w = (const float*)d_in[4];
  const float* out_b = (const float*)d_in[5];
  float* out = (float*)d_out;

  char* ws = (char*)d_ws;
  _Float16* h16   = (_Float16*)(ws);                     // 16 MB
  _Float16* wqkvT = (_Float16*)(ws + 16777216);          //  6 MB
  _Float16* woutT = (_Float16*)(ws + 23068672);          //  2 MB
  _Float16* q16   = (_Float16*)(ws + 25165824);          // 16 MB
  _Float16* k16   = (_Float16*)(ws + 41943040);          // 16 MB
  _Float16* vT16  = (_Float16*)(ws + 58720256);          // 16 MB
  _Float16* o16   = (_Float16*)(ws + 75497472);          // 16 MB (total ~88MB)

  convert_w_kernel<<<dim3((QKVN * C_ + C_ * C_) / 256), dim3(256), 0, stream>>>(
      qkv_w, out_w, wqkvT, woutT);
  ln_kernel<<<dim3(ROWS_), dim3(256), 0, stream>>>(x, ln_w, ln_b, h16);
  qkv_gemm_kernel<<<dim3(768), dim3(256), 0, stream>>>(h16, wqkvT, q16, k16, vT16);
  attn_kernel<<<dim3(512), dim3(256), 0, stream>>>(q16, k16, vT16, o16);
  out_gemm_kernel<<<dim3(256), dim3(256), 0, stream>>>(o16, woutT, out_b, x, out);
}
